// SpectralConv2d_83648783057486
// MI455X (gfx1250) — matrix operand order repaired and checked by reading, not compiled, not run on hardware
//
#include <hip/hip_runtime.h>
#include <stdint.h>
#include <math.h>

typedef __bf16 bf16_t;
typedef __bf16 v16bf __attribute__((ext_vector_type(16)));
typedef float  v8f   __attribute__((ext_vector_type(8)));
typedef unsigned int v4u __attribute__((ext_vector_type(4)));
typedef int    v4i   __attribute__((ext_vector_type(4)));
typedef int    v8i   __attribute__((ext_vector_type(8)));

#define EPSN 1e-12f

// ---------------- workspace layout (float indices) ----------------
#define WS_ZERO   0       // 64 floats: zero page (async-copy source for padded pixels)
#define WS_CNT    64      // grid-barrier ticket counter (u32)
#define WS_G      80      // 18 floats: G[h,w] complex accumulator
#define WS_BACC   128     // 512 floats: b accumulator
#define WS_D      640     // 6
#define WS_C      646     // 6
#define WS_SBAR   652     // 18
#define WS_ISIG   672     // 1 float: 1/sigma
#define WS_A      1024    // 512 floats: a (complex 256)
#define WS_B      1536    // 512 floats: b
#define WS_AACC   2048    // 512
#define WS_Q      2560    // 131072 floats: Q[o,i] complex
#define WS_BPRE   133632  // bf16[589824]: weights transposed to [cout][k]

// =================================================================
//  Cooperative power-iteration kernel (sigma) — 64 blocks x 256 thr
// =================================================================
__device__ inline void gridbar(unsigned* cnt, unsigned& epoch) {
  __syncthreads();
  __threadfence();
  if (threadIdx.x == 0) {
    atomicAdd(cnt, 1u);
    epoch += 64u;
    while (__hip_atomic_load(cnt, __ATOMIC_ACQUIRE, __HIP_MEMORY_SCOPE_AGENT) < epoch)
      __builtin_amdgcn_s_sleep(2);
  }
  __syncthreads();
}

__device__ inline float blockReduce(float v, float* red) {
  const int t = threadIdx.x;
  red[t] = v;
  __syncthreads();
  for (int s = 128; s > 0; s >>= 1) {
    if (t < s) red[t] += red[t + s];
    __syncthreads();
  }
  float r = red[0];
  __syncthreads();
  return r;
}

__global__ __launch_bounds__(256) void sigma_kernel(
    const float* __restrict__ w, const float* __restrict__ u1,
    const float* __restrict__ u2, const float* __restrict__ u3,
    float* ws)
{
  __shared__ float red[256];
  __shared__ float sred[18];
  unsigned* cnt = (unsigned*)(ws + WS_CNT);
  float* Gg   = ws + WS_G;
  float* bAcc = ws + WS_BACC;
  float* dv   = ws + WS_D;
  float* cv   = ws + WS_C;
  float* sb   = ws + WS_SBAR;
  float* av   = ws + WS_A;
  float* bv   = ws + WS_B;
  float* aAcc = ws + WS_AACC;
  float* Q    = ws + WS_Q;

  const int tid = threadIdx.x;
  const int blk = blockIdx.x;
  unsigned epoch = 0;

  if (blk == 0) {
    av[2*tid]   = u1[2*tid];   av[2*tid+1] = u1[2*tid+1];
    bv[2*tid]   = u2[2*tid];   bv[2*tid+1] = u2[2*tid+1];
    if (tid < 3) { cv[2*tid] = u3[2*tid]; cv[2*tid+1] = u3[2*tid+1]; }
  }
  gridbar(cnt, epoch);

  for (int it = 0; it <= 20; ++it) {
    // ---- P1: G[t] = sum_{o,i} P[o,i,t] * conj(a_o)*conj(b_i)
    float gr[9], gi[9];
    #pragma unroll
    for (int t = 0; t < 9; ++t) { gr[t] = 0.f; gi[t] = 0.f; }
    for (int j = 0; j < 4; ++j) {
      const int o = (blk << 2) + j;
      const int i = tid;
      const int idx = (o << 8) + i;
      const float ar = av[2*o], ai = av[2*o+1];
      const float br = bv[2*i], bi = bv[2*i+1];
      const float wr = ar*br - ai*bi;
      const float wi = -(ar*bi + ai*br);
      const float* P = w + (size_t)idx * 9;
      #pragma unroll
      for (int t = 0; t < 9; ++t) { const float p = P[t]; gr[t] += p*wr; gi[t] += p*wi; }
    }
    if (tid < 18) sred[tid] = 0.f;
    __syncthreads();
    #pragma unroll
    for (int t = 0; t < 9; ++t) { atomicAdd(&sred[2*t], gr[t]); atomicAdd(&sred[2*t+1], gi[t]); }
    __syncthreads();
    if (tid < 18) atomicAdd(&Gg[tid], sred[tid]);
    gridbar(cnt, epoch);

    if (it == 20) break;   // last pass only refreshes G for the final sigma

    // ---- P2: block0/thread0: d = norm(G c~), c = norm(G d~), sbar = conj(c x d)
    if (blk == 0 && tid == 0) {
      float Gr[9], Gi[9], cr[3], cI[3];
      #pragma unroll
      for (int t = 0; t < 9; ++t) { Gr[t] = Gg[2*t]; Gi[t] = Gg[2*t+1]; }
      #pragma unroll
      for (int h = 0; h < 3; ++h) { cr[h] = cv[2*h]; cI[h] = cv[2*h+1]; }
      float dr[3] = {0,0,0}, di[3] = {0,0,0};
      for (int h = 0; h < 3; ++h)
        for (int wq = 0; wq < 3; ++wq) {
          const int t = h*3 + wq;
          dr[wq] += Gr[t]*cr[h] + Gi[t]*cI[h];
          di[wq] += Gi[t]*cr[h] - Gr[t]*cI[h];
        }
      float nn = 0.f;
      for (int wq = 0; wq < 3; ++wq) nn += dr[wq]*dr[wq] + di[wq]*di[wq];
      nn = sqrtf(nn) + EPSN;
      for (int wq = 0; wq < 3; ++wq) {
        dr[wq] /= nn; di[wq] /= nn; dv[2*wq] = dr[wq]; dv[2*wq+1] = di[wq];
      }
      float er[3] = {0,0,0}, ei[3] = {0,0,0};
      for (int h = 0; h < 3; ++h)
        for (int wq = 0; wq < 3; ++wq) {
          const int t = h*3 + wq;
          er[h] += Gr[t]*dr[wq] + Gi[t]*di[wq];
          ei[h] += Gi[t]*dr[wq] - Gr[t]*di[wq];
        }
      nn = 0.f;
      for (int h = 0; h < 3; ++h) nn += er[h]*er[h] + ei[h]*ei[h];
      nn = sqrtf(nn) + EPSN;
      for (int h = 0; h < 3; ++h) { er[h] /= nn; ei[h] /= nn; cv[2*h] = er[h]; cv[2*h+1] = ei[h]; }
      for (int h = 0; h < 3; ++h)
        for (int wq = 0; wq < 3; ++wq) {
          const int t = h*3 + wq;
          sb[2*t]   = er[h]*dr[wq] - ei[h]*di[wq];
          sb[2*t+1] = -(er[h]*di[wq] + ei[h]*dr[wq]);
        }
    }
    gridbar(cnt, epoch);

    // ---- P3: Q[o,i] = sum_t P*sbar_t ; partial b_i += conj(a_o)*Q
    if (tid < 18) sred[tid] = sb[tid];
    __syncthreads();
    float pr = 0.f, pim = 0.f;
    for (int j = 0; j < 4; ++j) {
      const int o = (blk << 2) + j;
      const int idx = (o << 8) + tid;
      const float* P = w + (size_t)idx * 9;
      float qr = 0.f, qi = 0.f;
      #pragma unroll
      for (int t = 0; t < 9; ++t) { const float p = P[t]; qr += p*sred[2*t]; qi += p*sred[2*t+1]; }
      Q[2*idx] = qr; Q[2*idx+1] = qi;
      const float ar = av[2*o], ai = av[2*o+1];
      pr  += ar*qr + ai*qi;
      pim += ar*qi - ai*qr;
    }
    atomicAdd(&bAcc[2*tid],   pr);
    atomicAdd(&bAcc[2*tid+1], pim);
    gridbar(cnt, epoch);

    // ---- P4: block0 normalizes b
    if (blk == 0) {
      const float vr = bAcc[2*tid], vi = bAcc[2*tid+1];
      const float nn = sqrtf(blockReduce(vr*vr + vi*vi, red)) + EPSN;
      bv[2*tid] = vr / nn; bv[2*tid+1] = vi / nn;
    }
    gridbar(cnt, epoch);

    // ---- P5: a_o = sum_i conj(b_i)*Q[o,i] (block owns 4 o's -> plain store)
    for (int j = 0; j < 4; ++j) {
      const int o = (blk << 2) + j;
      const int idx = (o << 8) + tid;
      const float br = bv[2*tid], bi = bv[2*tid+1];
      const float qr = Q[2*idx], qi = Q[2*idx+1];
      const float sr = blockReduce(br*qr + bi*qi, red);
      const float si = blockReduce(br*qi - bi*qr, red);
      if (tid == 0) { aAcc[2*o] = sr; aAcc[2*o+1] = si; }
    }
    gridbar(cnt, epoch);

    // ---- P6: block0 normalizes a; re-zero G and bAcc for next iteration
    if (blk == 0) {
      const float vr = aAcc[2*tid], vi = aAcc[2*tid+1];
      const float nn = sqrtf(blockReduce(vr*vr + vi*vi, red)) + EPSN;
      av[2*tid] = vr / nn; av[2*tid+1] = vi / nn;
      if (tid < 18) Gg[tid] = 0.f;
      bAcc[2*tid] = 0.f; bAcc[2*tid+1] = 0.f;
    }
    gridbar(cnt, epoch);
  }

  // ---- final sigma = || sum_h G[h,w] conj(c_h) ||_2 ; store 1/sigma
  if (blk == 0 && tid == 0) {
    float Gr[9], Gi[9], cr[3], cI[3];
    #pragma unroll
    for (int t = 0; t < 9; ++t) { Gr[t] = Gg[2*t]; Gi[t] = Gg[2*t+1]; }
    #pragma unroll
    for (int h = 0; h < 3; ++h) { cr[h] = cv[2*h]; cI[h] = cv[2*h+1]; }
    float nn = 0.f;
    for (int wq = 0; wq < 3; ++wq) {
      float dr = 0.f, di = 0.f;
      for (int h = 0; h < 3; ++h) {
        const int t = h*3 + wq;
        dr += Gr[t]*cr[h] + Gi[t]*cI[h];
        di += Gi[t]*cr[h] - Gr[t]*cI[h];
      }
      nn += dr*dr + di*di;
    }
    const float sigma = sqrtf(nn);
    ws[WS_ISIG] = 1.0f / (sigma + 1e-30f);
    (void)dv;
  }
}

// =================================================================
//  Weight pre-transform: OIHW fp32 -> [cout][tap*256+cin] bf16
// =================================================================
__global__ __launch_bounds__(256) void wprep_kernel(const float* __restrict__ w,
                                                    bf16_t* __restrict__ Bpre) {
  const int idx = blockIdx.x * 256 + threadIdx.x;   // 589824 total
  const int o  = idx / 2304;
  const int k  = idx - o * 2304;
  const int t  = k >> 8;
  const int ci = k & 255;
  Bpre[idx] = (bf16_t)w[(size_t)o * 2304 + ci * 9 + t];
}

// =================================================================
//  Implicit-GEMM conv: M=131072, N=256, K=2304 (9 taps x 256 cin)
//  128x128 block tile, 8 waves (2M x 4N), K-step 32, bf16 WMMA.
//  A: global_load_async_to_lds_b128 (fp32, zero-page for padding)
//  B: TDM tensor_load_to_lds (2D tile 32x128, stride 2304, LDS pad)
// =================================================================
#define NSTAGE 72
#define AST 36   // A LDS row stride in floats  (144B = 9x16B)
#define BST 40   // B LDS row stride in bf16    (80B: 64B row + 16B TDM pad)

__global__ __launch_bounds__(256) void conv_kernel(
    const float* __restrict__ xin, const bf16_t* __restrict__ Bpre,
    const float* __restrict__ bias, const float* __restrict__ wsf,
    float* __restrict__ out)
{
  __shared__ float  As[2][128 * AST];
  __shared__ bf16_t Bs[2][128 * BST];

  const int tid  = threadIdx.x;
  const int bid  = blockIdx.x;
  const int m0   = (bid >> 1) << 7;
  const int n0   = (bid & 1) << 7;
  const int lane = tid & 31;
  const int wid  = tid >> 5;
  const int wm0  = (wid & 1) << 6;
  const int wn0  = (wid >> 1) << 5;

  const float* zpg = wsf + WS_ZERO;
  const int pbase = tid >> 3;
  const int q     = tid & 7;

  const int klo  = (lane >> 4) << 3;   // A K-run base (floats)
  const int mrow = lane & 15;
  const int kb   = (lane >> 4) << 3;   // B K-block base (bf16)
  const int ncol = lane & 15;

  v8f acc[4][2] = {};

  auto issueA = [&](int s, int buf) {
    const int tap = s >> 3;
    const int c0  = (s & 7) << 5;
    const int dy  = tap / 3 - 1;
    const int dx  = tap % 3 - 1;
    #pragma unroll
    for (int j = 0; j < 4; ++j) {
      const int p  = pbase + (j << 5);
      const int m  = m0 + p;
      const int iy = ((m >> 6) & 63) + dy;
      const int ix = (m & 63) + dx;
      const int im = m >> 12;
      const bool inb = ((unsigned)iy < 64u) & ((unsigned)ix < 64u);
      const float* src = inb
          ? (xin + (((size_t)((im << 6) + iy) << 6) + (size_t)ix) * 256 + c0 + (q << 2))
          : zpg;
      const uint32_t ldsa = (uint32_t)(uintptr_t)&As[buf][p * AST + (q << 2)];
      asm volatile("global_load_async_to_lds_b128 %0, %1, off"
                   :: "v"(ldsa), "v"(src) : "memory");
    }
  };

  auto issueB = [&](int s, int buf) {
    if (wid != 0) return;
    const int tap = s >> 3;
    const int k0  = tap * 256 + ((s & 7) << 5);
    const uint64_t ga  = (uint64_t)(uintptr_t)(Bpre + (size_t)n0 * 2304 + k0);
    const uint32_t ldb = (uint32_t)(uintptr_t)&Bs[buf][0];
    // D# group0: count=1, lds_addr, 57-bit global addr, type=2
    v4u g0 = { 1u, ldb, (uint32_t)ga, (uint32_t)(ga >> 32) | (2u << 30) };
    // D# group1: data_size=2B, pad_enable, pad_interval=16DW, pad_amount=4DW,
    //            tensor_dim0=2304, tensor_dim1=256, tile_dim0=32, tile_dim1=128,
    //            tensor_dim0_stride=2304
    v8i g1 = { (int)((1u << 16) | (1u << 20) | (3u << 22) | (3u << 25)),
               (int)(2304u << 16),
               (int)(256u << 16),
               (int)(32u << 16),
               128,
               2304,
               0, 0 };
    v4i g2 = { 0, 0, 0, 0 };
    v4i g3 = { 0, 0, 0, 0 };
    v8i g4 = { 0, 0, 0, 0, 0, 0, 0, 0 };
    __builtin_amdgcn_tensor_load_to_lds(g0, g1, g2, g3, g4, 0);
  };

  issueA(0, 0);
  issueB(0, 0);

  for (int s = 0; s < NSTAGE; ++s) {
    const int buf = s & 1;
    if (s + 1 < NSTAGE) {
      issueA(s + 1, buf ^ 1);
      issueB(s + 1, buf ^ 1);
      asm volatile("s_wait_asynccnt 4" ::: "memory");
      if (wid == 0) __builtin_amdgcn_s_wait_tensorcnt(1);
    } else {
      asm volatile("s_wait_asynccnt 0" ::: "memory");
      if (wid == 0) __builtin_amdgcn_s_wait_tensorcnt(0);
    }
    __syncthreads();

    // A fragments: convert fp32 -> bf16 while building per-lane rows
    v16bf aF[4];
    #pragma unroll
    for (int f = 0; f < 4; ++f) {
      const float* ap = &As[buf][(wm0 + (f << 4) + mrow) * AST + klo];
      const float4 r0 = *(const float4*)(ap);
      const float4 r1 = *(const float4*)(ap + 4);
      const float4 r2 = *(const float4*)(ap + 16);
      const float4 r3 = *(const float4*)(ap + 20);
      v16bf a;
      a[0]=(bf16_t)r0.x;  a[1]=(bf16_t)r0.y;  a[2]=(bf16_t)r0.z;  a[3]=(bf16_t)r0.w;
      a[4]=(bf16_t)r1.x;  a[5]=(bf16_t)r1.y;  a[6]=(bf16_t)r1.z;  a[7]=(bf16_t)r1.w;
      a[8]=(bf16_t)r2.x;  a[9]=(bf16_t)r2.y;  a[10]=(bf16_t)r2.z; a[11]=(bf16_t)r2.w;
      a[12]=(bf16_t)r3.x; a[13]=(bf16_t)r3.y; a[14]=(bf16_t)r3.z; a[15]=(bf16_t)r3.w;
      aF[f] = a;
    }
    // B fragments: two 8-bf16 K runs per lane (k=kb..kb+7, kb+16..kb+23)
    v16bf bF[2];
    #pragma unroll
    for (int g = 0; g < 2; ++g) {
      const bf16_t* bp = &Bs[buf][(wn0 + (g << 4) + ncol) * BST + kb];
      union { v16bf v; uint4 u[2]; } bu;
      bu.u[0] = *(const uint4*)(bp);
      bu.u[1] = *(const uint4*)(bp + 16);
      bF[g] = bu.v;
    }
    #pragma unroll
    for (int f = 0; f < 4; ++f)
      #pragma unroll
      for (int g = 0; g < 2; ++g)
        acc[f][g] = __builtin_amdgcn_wmma_f32_16x16x32_bf16(
            false, aF[f], false, bF[g], (short)0, acc[f][g], false, false);

    __syncthreads();
  }

  // epilogue: y = acc * (1/sigma) + bias
  const float isg = wsf[WS_ISIG];
  #pragma unroll
  for (int g = 0; g < 2; ++g) {
    const int n = n0 + wn0 + (g << 4) + ncol;
    const float bb = bias[n];
    #pragma unroll
    for (int f = 0; f < 4; ++f) {
      const int mb = m0 + wm0 + (f << 4) + ((lane >> 4) << 3);
      #pragma unroll
      for (int r = 0; r < 8; ++r) {
        out[(size_t)(mb + r) * 256 + n] = acc[f][g][r] * isg + bb;
      }
    }
  }
}

// =================================================================
extern "C" void kernel_launch(void* const* d_in, const int* in_sizes, int n_in,
                              void* d_out, int out_size, void* d_ws, size_t ws_size,
                              hipStream_t stream) {
  (void)in_sizes; (void)n_in; (void)out_size; (void)ws_size;
  const float* x  = (const float*)d_in[0];
  const float* w  = (const float*)d_in[1];
  const float* b  = (const float*)d_in[2];
  const float* u1 = (const float*)d_in[3];
  const float* u2 = (const float*)d_in[4];
  const float* u3 = (const float*)d_in[5];
  float* out = (float*)d_out;
  float* ws  = (float*)d_ws;

  // zero: zero-page, barrier counter, G accumulator, bAcc
  (void)hipMemsetAsync(d_ws, 0, 4096, stream);
  sigma_kernel<<<64, 256, 0, stream>>>(w, u1, u2, u3, ws);
  bf16_t* Bpre = (bf16_t*)(ws + WS_BPRE);
  wprep_kernel<<<2304, 256, 0, stream>>>(w, Bpre);
  conv_kernel<<<2048, 256, 0, stream>>>(x, Bpre, b, ws, out);
}
